// MultiheadAttention_11802570129446
// MI455X (gfx1250) — compile-verified, matrix operand order checked by reading, not run on hardware
//
#include <hip/hip_runtime.h>
#include <hip/hip_bf16.h>
#include <cstdint>

// MHA forward for MI455X (gfx1250, wave32, WMMA + Tensor Data Mover).
// Pipeline: f32->f16 convert -> QKV projections (WMMA) -> flash attention
// (WMMA + online softmax) -> output projection (WMMA, f32 out).
// GEMM math on v_wmma_f32_16x16x32_f16 (f32 accum); tiles staged into LDS by
// the TDM (tensor_load_to_lds, double-buffered) when the builtin is available.

#define S_LEN 2048
#define E_DIM 1024
#define H_NUM 16
#define D_HEAD 64
#define B_NUM 2
#define M_ROWS (B_NUM * S_LEN)   // 4096

typedef __attribute__((ext_vector_type(16))) _Float16 v16h;
typedef __attribute__((ext_vector_type(8)))  float    v8f;

#if defined(__has_builtin)
#if __has_builtin(__builtin_amdgcn_tensor_load_to_lds) && \
    __has_builtin(__builtin_amdgcn_s_wait_tensorcnt)
#define USE_TDM 1
#endif
#endif
#ifndef USE_TDM
#define USE_TDM 0
#endif

union F16x16 {
    v16h v;
    uint4 q[2];
};

__device__ __forceinline__ v8f wmma_f16(v16h a, v16h b, v8f c) {
    // (neg_a, A, neg_b, B, c_mod, C, reuse_a, reuse_b)
    return __builtin_amdgcn_wmma_f32_16x16x32_f16(false, a, false, b, (short)0, c,
                                                  false, false);
}

// Load a 16x32 f16 fragment in A/B layout:
// lane L -> row (L%16), 16 contiguous halves starting at colBase + (L/16)*16.
__device__ __forceinline__ v16h ldfrag(const _Float16* base, int ld, int colBase) {
    int lane = threadIdx.x & 31;
    const _Float16* p = base + (lane & 15) * ld + colBase + ((lane >> 4) << 4);
    F16x16 f;
    f.q[0] = *(const uint4*)p;
    f.q[1] = *(const uint4*)(p + 8);
    return f.v;
}

#if USE_TDM
typedef __attribute__((ext_vector_type(4))) unsigned int u32x4;
typedef __attribute__((ext_vector_type(8))) int          i32x8;
typedef __attribute__((ext_vector_type(4))) int          i32x4;

// Issue one TDM 2D tile load (f16 elements) global -> LDS with LDS row padding.
// row_elems halves per row, `rows` rows, global row stride in halves.
// pad_icode: pad interval = 2<<code DWORDs; pad_acode: pad amount = code+1 DWORDs.
// Resulting LDS row stride = row_elems + (pad_acode+1)*2 halves.
__device__ __forceinline__ void tdm_load_tile_f16(
    void* lds_dst, const _Float16* gsrc, uint32_t row_elems, uint32_t rows,
    uint64_t row_stride_elems, uint32_t pad_icode, uint32_t pad_acode) {
    const uint64_t ga = (uint64_t)(uintptr_t)gsrc;    // global byte address
    const uint32_t la = (uint32_t)(uintptr_t)lds_dst; // LDS byte address (addr[31:0])
    u32x4 g0;
    g0.x = 1u;                                        // count=1 (valid user D#)
    g0.y = la;                                        // lds_addr
    g0.z = (uint32_t)ga;                              // global_addr[31:0]
    g0.w = (uint32_t)((ga >> 32) & 0x01ffffffu) | 0x80000000u; // ga[56:32] | type=2
    const uint32_t flags = (1u << 16)                 // data_size = 2 bytes
                         | (1u << 20)                 // pad_enable
                         | (pad_icode << 22) | (pad_acode << 25);
    i32x8 g1;
    g1[0] = (int)flags;
    g1[1] = (int)((row_elems & 0xffffu) << 16);       // tensor_dim0[15:0] (barrier addr 0)
    g1[2] = (int)(((row_elems >> 16) & 0xffffu) | ((rows & 0xffffu) << 16));
    g1[3] = (int)(((rows >> 16) & 0xffffu) | ((row_elems & 0xffffu) << 16)); // tile_dim0
    g1[4] = (int)(rows & 0xffffu);                    // tile_dim1 (tile_dim2 = 0)
    g1[5] = (int)(uint32_t)(row_stride_elems & 0xffffffffu);   // tensor_dim0_stride lo
    g1[6] = (int)(uint32_t)((row_stride_elems >> 32) & 0xffffu);
    g1[7] = 0;
    const i32x4 z4 = {0, 0, 0, 0};                    // groups 2/3 unused (2D)
    const i32x8 z8 = {0, 0, 0, 0, 0, 0, 0, 0};
    // amdgpu-toolchain (clang-23) 6-arg form
    __builtin_amdgcn_tensor_load_to_lds(g0, g1, z4, z4, z8, 0);
}
#endif

// ---------------------------------------------------------------- convert
__global__ void cvt_f32_f16_kernel(const float* __restrict__ src,
                                   _Float16* __restrict__ dst, int n) {
    int i = (blockIdx.x * 256 + threadIdx.x) * 4;
    if (i < n) {
        float4 v = *(const float4*)(src + i);
        union { _Float16 h[4]; uint2 u; } pk;
        pk.h[0] = (_Float16)v.x; pk.h[1] = (_Float16)v.y;
        pk.h[2] = (_Float16)v.z; pk.h[3] = (_Float16)v.w;
        *(uint2*)(dst + i) = pk.u;
    }
}

// ---------------------------------------------------------------- projection
// Y[m,n] = sum_k X[m,k] * W[n,k] + bias[n]   (M=4096, N=1024, K=1024)
// mode 0: Y -> f16 [B,H,S,D]; mode 1: Y -> f16 [B,H,D,S]; mode 2: Y -> f32 [M,1024]
__global__ __launch_bounds__(256)
void proj_kernel(const _Float16* __restrict__ X, const _Float16* __restrict__ W,
                 const float* __restrict__ bias, _Float16* __restrict__ outH,
                 float* __restrict__ outF, int mode) {
    const int m0 = blockIdx.x * 128;
    const int n0 = blockIdx.y * 128;
    const int t = threadIdx.x;
    const int w = t >> 5;
    const int lane = t & 31;

    v8f C[8];
#pragma unroll
    for (int s = 0; s < 8; ++s) C[s] = {};

#if USE_TDM
    // double-buffered TDM staging: wave 0 drives the DMA, all waves compute
    __shared__ __attribute__((aligned(16))) _Float16 Xs[2][128 * 40];
    __shared__ __attribute__((aligned(16))) _Float16 Ws[2][128 * 40];
    if (t < 32) {
        tdm_load_tile_f16(Xs[0], X + (size_t)m0 * E_DIM, 32, 128, E_DIM, 3, 3);
        tdm_load_tile_f16(Ws[0], W + (size_t)n0 * E_DIM, 32, 128, E_DIM, 3, 3);
    }
    for (int kt = 0; kt < 32; ++kt) {
        const int cur = kt & 1;
        if (t < 32) {
            if (kt + 1 < 32) {
                const int k1 = (kt + 1) * 32;
                tdm_load_tile_f16(Xs[cur ^ 1], X + (size_t)m0 * E_DIM + k1,
                                  32, 128, E_DIM, 3, 3);
                tdm_load_tile_f16(Ws[cur ^ 1], W + (size_t)n0 * E_DIM + k1,
                                  32, 128, E_DIM, 3, 3);
                __builtin_amdgcn_s_wait_tensorcnt(2);   // current pair done
            } else {
                __builtin_amdgcn_s_wait_tensorcnt(0);
            }
        }
        __syncthreads();
        v16h A = ldfrag(&Xs[cur][w * 16 * 40], 40, 0);
#pragma unroll
        for (int s = 0; s < 8; ++s) {
            v16h Bf = ldfrag(&Ws[cur][s * 16 * 40], 40, 0);
            C[s] = wmma_f16(A, Bf, C[s]);
        }
        __syncthreads();
    }
#else
    __shared__ __attribute__((aligned(16))) _Float16 Xs[128 * 40];
    __shared__ __attribute__((aligned(16))) _Float16 Ws[128 * 40];
    const int row = t >> 1;
    const int chunk = (t & 1) << 4;
    for (int kt = 0; kt < 32; ++kt) {
        const int k0 = kt * 32;
        *(uint4*)&Xs[row * 40 + chunk] =
            *(const uint4*)&X[(size_t)(m0 + row) * E_DIM + k0 + chunk];
        *(uint4*)&Xs[row * 40 + chunk + 8] =
            *(const uint4*)&X[(size_t)(m0 + row) * E_DIM + k0 + chunk + 8];
        *(uint4*)&Ws[row * 40 + chunk] =
            *(const uint4*)&W[(size_t)(n0 + row) * E_DIM + k0 + chunk];
        *(uint4*)&Ws[row * 40 + chunk + 8] =
            *(const uint4*)&W[(size_t)(n0 + row) * E_DIM + k0 + chunk + 8];
        __syncthreads();
        v16h A = ldfrag(Xs + w * 16 * 40, 40, 0);
#pragma unroll
        for (int s = 0; s < 8; ++s) {
            v16h Bf = ldfrag(Ws + s * 16 * 40, 40, 0);
            C[s] = wmma_f16(A, Bf, C[s]);
        }
        __syncthreads();
    }
#endif

    // epilogue: bias + scatter in the requested layout
#pragma unroll
    for (int s = 0; s < 8; ++s) {
        const int n = n0 + s * 16 + (lane & 15);
        const float bval = bias[n];
#pragma unroll
        for (int r = 0; r < 8; ++r) {
            const int m = m0 + w * 16 + r + ((lane >> 4) << 3);
            const float val = C[s][r] + bval;
            if (mode == 2) {
                outF[(size_t)m * E_DIM + n] = val;
            } else {
                const int bb = m >> 11, sidx = m & (S_LEN - 1);
                const int hh = n >> 6, dd = n & 63;
                const size_t idx =
                    (mode == 0)
                        ? (((size_t)(bb * H_NUM + hh) * S_LEN + sidx) * D_HEAD + dd)
                        : (((size_t)(bb * H_NUM + hh) * D_HEAD + dd) * S_LEN + sidx);
                outH[idx] = (_Float16)val;
            }
        }
    }
}

// ---------------------------------------------------------------- attention
// One block = (b,h) x 128 query rows; each of 8 waves owns a 16-row q tile.
__global__ __launch_bounds__(256)
void attn_kernel(const _Float16* __restrict__ Qh, const _Float16* __restrict__ Kh,
                 const _Float16* __restrict__ Vt, _Float16* __restrict__ ctx) {
    __shared__ __attribute__((aligned(16))) _Float16 Ps[8 * 16 * 40];  // per-wave P

    const int bh = blockIdx.x >> 4;
    const int qb = blockIdx.x & 15;
    const int b = bh >> 4;
    const int h = bh & 15;
    const int t = threadIdx.x;
    const int w = t >> 5;
    const int lane = t & 31;
    const int q0 = qb * 128 + w * 16;

    const _Float16* Qb = Qh + (size_t)bh * S_LEN * D_HEAD;
    const _Float16* Kb = Kh + (size_t)bh * S_LEN * D_HEAD;
    const _Float16* Vb = Vt + (size_t)bh * D_HEAD * S_LEN;

    const v16h A0 = ldfrag(Qb + q0 * D_HEAD, D_HEAD, 0);
    const v16h A1 = ldfrag(Qb + q0 * D_HEAD, D_HEAD, 32);

    v8f O[4];
#pragma unroll
    for (int f = 0; f < 4; ++f) O[f] = {};
    float m_run[8], l_run[8];
#pragma unroll
    for (int r = 0; r < 8; ++r) { m_run[r] = -INFINITY; l_run[r] = 0.0f; }

    const int ktiles = qb * 4 + 4;

#if USE_TDM
    __shared__ __attribute__((aligned(16))) _Float16 Ks[2][32 * 72];  // [key][d]
    __shared__ __attribute__((aligned(16))) _Float16 Vs[2][64 * 40];  // [d][key]
    if (t < 32) {
        tdm_load_tile_f16(Ks[0], Kb, 64, 32, D_HEAD, 4, 3);
        tdm_load_tile_f16(Vs[0], Vb, 32, 64, S_LEN, 3, 3);
    }
#else
    __shared__ __attribute__((aligned(16))) _Float16 Ks[1][32 * 72];
    __shared__ __attribute__((aligned(16))) _Float16 Vs[1][64 * 40];
    const int krow = t >> 3, kcol = (t & 7) << 3;
    const int vrow = t >> 2, vcol = (t & 3) << 3;
#endif

    for (int kt = 0; kt < ktiles; ++kt) {
        const int kbase = kt * 32;
#if USE_TDM
        const int cur = kt & 1;
        if (t < 32) {
            if (kt + 1 < ktiles) {
                tdm_load_tile_f16(Ks[cur ^ 1], Kb + (size_t)(kbase + 32) * D_HEAD,
                                  64, 32, D_HEAD, 4, 3);
                tdm_load_tile_f16(Vs[cur ^ 1], Vb + kbase + 32,
                                  32, 64, S_LEN, 3, 3);
                __builtin_amdgcn_s_wait_tensorcnt(2);
            } else {
                __builtin_amdgcn_s_wait_tensorcnt(0);
            }
        }
#else
        const int cur = 0;
        *(uint4*)&Ks[0][krow * 72 + kcol] =
            *(const uint4*)&Kb[(size_t)(kbase + krow) * D_HEAD + kcol];
        *(uint4*)&Vs[0][vrow * 40 + vcol] =
            *(const uint4*)&Vb[(size_t)vrow * S_LEN + kbase + vcol];
        if (kt + 1 < ktiles) {
            __builtin_prefetch(&Kb[(size_t)(kbase + 32 + krow) * D_HEAD + kcol], 0, 0);
            __builtin_prefetch(&Vb[(size_t)vrow * S_LEN + kbase + 32 + vcol], 0, 0);
        }
#endif
        __syncthreads();

        if (kbase <= q0 + 15) {
            // ---- scores: 16q x 32k, two 16-key groups
            float sc[2][8];
#pragma unroll
            for (int g = 0; g < 2; ++g) {
                v8f c = {};
                c = wmma_f16(A0, ldfrag(&Ks[cur][g * 16 * 72], 72, 0), c);
                c = wmma_f16(A1, ldfrag(&Ks[cur][g * 16 * 72], 72, 32), c);
                const int colk = kbase + g * 16 + (lane & 15);
#pragma unroll
                for (int r = 0; r < 8; ++r) {
                    const int qrow = q0 + r + ((lane >> 4) << 3);
                    sc[g][r] = (colk <= qrow) ? c[r] * 0.125f : -1.0e9f;
                }
            }
            // ---- online softmax (row reductions across 16-lane groups)
            float mt[8];
#pragma unroll
            for (int r = 0; r < 8; ++r) mt[r] = fmaxf(sc[0][r], sc[1][r]);
#pragma unroll
            for (int off = 1; off < 16; off <<= 1)
#pragma unroll
                for (int r = 0; r < 8; ++r)
                    mt[r] = fmaxf(mt[r], __shfl_xor(mt[r], off, 32));
            float al[8];
#pragma unroll
            for (int r = 0; r < 8; ++r) {
                const float mn = fmaxf(m_run[r], mt[r]);
                al[r] = __expf(m_run[r] - mn);
                m_run[r] = mn;
            }
            float ls[8];
#pragma unroll
            for (int r = 0; r < 8; ++r) {
                sc[0][r] = __expf(sc[0][r] - m_run[r]);
                sc[1][r] = __expf(sc[1][r] - m_run[r]);
                ls[r] = sc[0][r] + sc[1][r];
            }
#pragma unroll
            for (int off = 1; off < 16; off <<= 1)
#pragma unroll
                for (int r = 0; r < 8; ++r) ls[r] += __shfl_xor(ls[r], off, 32);
#pragma unroll
            for (int r = 0; r < 8; ++r) l_run[r] = l_run[r] * al[r] + ls[r];
#pragma unroll
            for (int f = 0; f < 4; ++f)
#pragma unroll
                for (int r = 0; r < 8; ++r) O[f][r] *= al[r];

            // ---- restage P as f16 in A-layout via per-wave LDS scratch
            _Float16* pw = Ps + w * 16 * 40;
#pragma unroll
            for (int g = 0; g < 2; ++g)
#pragma unroll
                for (int r = 0; r < 8; ++r)
                    pw[(r + ((lane >> 4) << 3)) * 40 + g * 16 + (lane & 15)] =
                        (_Float16)sc[g][r];
            const v16h Ap = ldfrag(pw, 40, 0);
#pragma unroll
            for (int f = 0; f < 4; ++f)
                O[f] = wmma_f16(Ap, ldfrag(&Vs[cur][f * 16 * 40], 40, 0), O[f]);
        }
        __syncthreads();
    }

    // ---- normalize and store ctx f16 [B,S,E]
#pragma unroll
    for (int r = 0; r < 8; ++r) l_run[r] = 1.0f / l_run[r];
#pragma unroll
    for (int f = 0; f < 4; ++f)
#pragma unroll
        for (int r = 0; r < 8; ++r) {
            const int m = q0 + r + ((lane >> 4) << 3);
            ctx[((size_t)(b * S_LEN + m)) * E_DIM + h * 64 + f * 16 + (lane & 15)] =
                (_Float16)(O[f][r] * l_run[r]);
        }
}

// ---------------------------------------------------------------- launch
extern "C" void kernel_launch(void* const* d_in, const int* in_sizes, int n_in,
                              void* d_out, int out_size, void* d_ws, size_t ws_size,
                              hipStream_t stream) {
    (void)in_sizes; (void)n_in; (void)out_size; (void)ws_size;
    const float* q  = (const float*)d_in[0];
    const float* k  = (const float*)d_in[1];
    const float* v  = (const float*)d_in[2];
    // d_in[3] = causal mask (static, unused)
    const float* Wq = (const float*)d_in[4];
    const float* bq = (const float*)d_in[5];
    const float* Wk = (const float*)d_in[6];
    const float* bk = (const float*)d_in[7];
    const float* Wv = (const float*)d_in[8];
    const float* bv = (const float*)d_in[9];
    const float* Wo = (const float*)d_in[10];
    const float* bo = (const float*)d_in[11];
    float* out = (float*)d_out;

    char* ws = (char*)d_ws;
    const size_t SZ_X = (size_t)M_ROWS * E_DIM * 2;   // 8 MiB
    const size_t SZ_W = (size_t)E_DIM * E_DIM * 2;    // 2 MiB
    _Float16* XQ  = (_Float16*)(ws);
    _Float16* XK  = (_Float16*)(ws + SZ_X);
    _Float16* XV  = (_Float16*)(ws + 2 * SZ_X);
    _Float16* WQh = (_Float16*)(ws + 3 * SZ_X);
    _Float16* WKh = (_Float16*)(ws + 3 * SZ_X + SZ_W);
    _Float16* WVh = (_Float16*)(ws + 3 * SZ_X + 2 * SZ_W);
    _Float16* WOh = (_Float16*)(ws + 3 * SZ_X + 3 * SZ_W);
    _Float16* QH  = (_Float16*)(ws + 3 * SZ_X + 4 * SZ_W);
    _Float16* KH  = (_Float16*)(ws + 4 * SZ_X + 4 * SZ_W);
    _Float16* VT  = (_Float16*)(ws + 5 * SZ_X + 4 * SZ_W);
    _Float16* CTX = (_Float16*)(ws + 6 * SZ_X + 4 * SZ_W);

    const int NX = M_ROWS * E_DIM;
    const int NW = E_DIM * E_DIM;
    cvt_f32_f16_kernel<<<NX / 1024, 256, 0, stream>>>(q, XQ, NX);
    cvt_f32_f16_kernel<<<NX / 1024, 256, 0, stream>>>(k, XK, NX);
    cvt_f32_f16_kernel<<<NX / 1024, 256, 0, stream>>>(v, XV, NX);
    cvt_f32_f16_kernel<<<NW / 1024, 256, 0, stream>>>(Wq, WQh, NW);
    cvt_f32_f16_kernel<<<NW / 1024, 256, 0, stream>>>(Wk, WKh, NW);
    cvt_f32_f16_kernel<<<NW / 1024, 256, 0, stream>>>(Wv, WVh, NW);
    cvt_f32_f16_kernel<<<NW / 1024, 256, 0, stream>>>(Wo, WOh, NW);

    dim3 pg(M_ROWS / 128, E_DIM / 128);  // 32 x 8
    proj_kernel<<<pg, 256, 0, stream>>>(XQ, WQh, bq, QH, nullptr, 0);
    proj_kernel<<<pg, 256, 0, stream>>>(XK, WKh, bk, KH, nullptr, 0);
    proj_kernel<<<pg, 256, 0, stream>>>(XV, WVh, bv, VT, nullptr, 1);

    attn_kernel<<<B_NUM * H_NUM * (S_LEN / 128), 256, 0, stream>>>(QH, KH, VT, CTX);

    proj_kernel<<<pg, 256, 0, stream>>>(CTX, WOh, bo, nullptr, out, 2);
}
